// OneLayerTransformer_72593537237366
// MI455X (gfx1250) — matrix operand order repaired and checked by reading, not compiled, not run on hardware
//
#include <hip/hip_runtime.h>

// ---------------------------------------------------------------------------
// One-layer transformer forward for MI455X (gfx1250).
// bf16 WMMA for all matmuls, TDM + async global->LDS staging in the GEMM.
// ---------------------------------------------------------------------------

#define B_ 2
#define T_ 2048
#define D_ 1024
#define H_ 16
#define DH_ 64
#define V_ 50257
#define M_ (B_ * T_)   // 4096 rows of (b,t)

typedef __bf16    v16bf __attribute__((ext_vector_type(16)));
typedef float     v8f   __attribute__((ext_vector_type(8)));
typedef unsigned  v4u   __attribute__((ext_vector_type(4)));
typedef int       v8i   __attribute__((ext_vector_type(8)));
typedef int       v4i   __attribute__((ext_vector_type(4)));

union Frag16 {            // one WMMA 16-bit operand fragment (8 VGPRs / lane)
    v16bf    v;
    unsigned u[8];
};

__device__ __forceinline__ v8f wmma_bf16(const Frag16& a, const Frag16& b, v8f c) {
    return __builtin_amdgcn_wmma_f32_16x16x32_bf16(
        false, a.v, false, b.v, (short)0, c, false, false);
}

// A-fragment K index for VGPR v, lane-half hi (contiguous k order, matches B frags)
__device__ __forceinline__ int a_kidx(int v, int hi) {
    return 2 * v + 16 * hi;
}

// Low 32 bits of a flat LDS pointer are the LDS byte address (ISA 10.2).
__device__ __forceinline__ unsigned lds_offset(const void* p) {
    return (unsigned)(size_t)p;
}

// ---- gfx1250 async global->LDS (ASYNCcnt) --------------------------------
__device__ __forceinline__ void async_load_b128_to_lds(const __bf16* g, __bf16* l) {
#if __has_builtin(__builtin_amdgcn_global_load_async_to_lds_b128)
    __builtin_amdgcn_global_load_async_to_lds_b128(
        (__attribute__((address_space(1))) v4i*)(void*)g,
        (__attribute__((address_space(3))) v4i*)(void*)l, 0, 0);
#else
    unsigned loff = lds_offset(l);
    unsigned long long ga = (unsigned long long)(size_t)g;
    asm volatile("global_load_async_to_lds_b128 %0, %1, off"
                 :: "v"(loff), "v"(ga) : "memory");
#endif
}

__device__ __forceinline__ void wait_asynccnt0() {
#if __has_builtin(__builtin_amdgcn_s_wait_asynccnt)
    __builtin_amdgcn_s_wait_asynccnt(0);
#else
    asm volatile("s_wait_asynccnt 0x0" ::: "memory");
#endif
}

// ---- gfx1250 Tensor Data Mover: 2D tile (rows x elems) global->LDS -------
// Descriptor per CDNA5 ISA (D# groups 0/1).
__device__ __forceinline__ void tdm_load_2d_bf16(const __bf16* gaddr, unsigned ldsoff,
                                                 int elems_x, int rows,
                                                 long long row_stride_elems) {
    unsigned long long ga = (unsigned long long)(size_t)gaddr;
    v4u g0;
    g0.x = 1u;                                              // count=1 (valid D#)
    g0.y = ldsoff;                                          // lds_addr (bytes)
    g0.z = (unsigned)ga;                                    // global_addr[31:0]
    g0.w = (unsigned)((ga >> 32) & 0x1FFFFFFu) | (2u << 30);// addr[56:32], type=2
    v8i g1;
    g1[0] = 1 << 16;                                        // data_size=1 (2B)
    g1[1] = (elems_x & 0xFFFF) << 16;                       // tensor_dim0 lo
    g1[2] = ((elems_x >> 16) & 0xFFFF) | ((rows & 0xFFFF) << 16); // dim0 hi|dim1 lo
    g1[3] = ((rows >> 16) & 0xFFFF) | ((elems_x & 0xFFFF) << 16); // dim1 hi|tile_dim0
    g1[4] = rows & 0xFFFF;                                  // tile_dim1, tile_dim2=0
    g1[5] = (int)(row_stride_elems & 0xFFFFFFFFll);         // dim0_stride lo
    g1[6] = (int)((row_stride_elems >> 32) & 0xFFFFll);     // dim0_stride hi
    g1[7] = 0;                                              // dim1_stride=0
    v4i z4 = {0, 0, 0, 0};
#if defined(__clang_major__) && __clang_major__ >= 23
    v8i z8 = {0, 0, 0, 0, 0, 0, 0, 0};
    __builtin_amdgcn_tensor_load_to_lds(g0, g1, z4, z4, z8, 0);
#else
    __builtin_amdgcn_tensor_load_to_lds(g0, g1, z4, z4, 0);
#endif
}

// ---------------------------------------------------------------------------
// 1) f32 [K,N] -> bf16 [N,K] tiled transpose-convert (makes B-frags b32 loads)
// ---------------------------------------------------------------------------
__global__ __launch_bounds__(256) void transpose_convert_kernel(
    const float* __restrict__ W, __bf16* __restrict__ Wt, int K, int N) {
    __shared__ float tile[32][33];
    int k0 = blockIdx.y * 32, n0 = blockIdx.x * 32;
    int tx = threadIdx.x, ty = threadIdx.y;
    #pragma unroll
    for (int i = ty; i < 32; i += 8) {
        int k = k0 + i, n = n0 + tx;
        tile[i][tx] = (k < K && n < N) ? W[(size_t)k * N + n] : 0.f;
    }
    __syncthreads();
    #pragma unroll
    for (int i = ty; i < 32; i += 8) {
        int n = n0 + i, k = k0 + tx;
        if (n < N && k < K) Wt[(size_t)n * K + k] = (__bf16)tile[tx][i];
    }
}

// ---------------------------------------------------------------------------
// 2) Embedding: h = W_E[x] + W_pos  (f32 for residual, bf16 for GEMM A)
// ---------------------------------------------------------------------------
__global__ __launch_bounds__(256) void embed_kernel(
    const int* __restrict__ x, const float* __restrict__ W_E,
    const float* __restrict__ W_pos, float* __restrict__ hf,
    __bf16* __restrict__ hb) {
    int idx = blockIdx.x * 256 + threadIdx.x;       // over M_*D_
    int m = idx >> 10, d = idx & 1023;
    int t = m & (T_ - 1);
    float v = W_E[(size_t)x[m] * D_ + d] + W_pos[(size_t)t * D_ + d];
    hf[idx] = v;
    hb[idx] = (__bf16)v;
}

// ---------------------------------------------------------------------------
// 3) bf16 WMMA GEMM: C[M,N] = A[M,K] * Bt[N,K]^T, f32 accumulate.
//    WG = 128 threads (4 waves). WG tile 128(M) x 64(N), K-step 32.
//    Double-buffered LDS: A tile via TDM, B tile via async b128 copies.
// ---------------------------------------------------------------------------
enum { EPI_QHEAD = 0, EPI_VHEAD = 1, EPI_OPROJ = 2, EPI_UNEMB = 3 };

template <int EPI>
__global__ __launch_bounds__(128) void gemm_kernel(
    const __bf16* __restrict__ A, const __bf16* __restrict__ Bt,
    int M, int N, int K, void* __restrict__ out,
    const float* __restrict__ extra) {   // residual (OPROJ) or bias (UNEMB)
    __shared__ __bf16 atile[2][128 * 32];    // 2 x 8 KB
    __shared__ __bf16 btile[2][64 * 32];     // 2 x 4 KB
    int tid = threadIdx.x;
    int lane = tid & 31, wave = tid >> 5;
    int hi = lane >> 4, lo = lane & 15;
    int n0 = blockIdx.x * 64;
    int row0 = blockIdx.y * 128;
    int mw = wave * 32;                      // this wave's rows inside the tile

    auto issue_stage = [&](int ks, int buf) {
        // B tile: 64 rows x 32 bf16; 16B chunks, 2 per thread, async DMA.
        #pragma unroll
        for (int c2 = 0; c2 < 2; ++c2) {
            int ci = tid * 2 + c2;           // 0..255
            int r = ci >> 2, part = ci & 3;
            int n = n0 + r;
            int nn = (n < N) ? n : (N - 1);  // clamp tail (masked at store)
            const __bf16* g = Bt + (size_t)nn * K + ks + part * 8;
            async_load_b128_to_lds(g, &btile[buf][r * 32 + part * 8]);
            __builtin_prefetch(g + 64, 0, 1);          // 2 k-steps ahead
        }
        // A tile: 128 rows x 32 bf16 via Tensor Data Mover (one wave issues).
        if (wave == 0)
            tdm_load_2d_bf16(A + (size_t)row0 * K + ks,
                             lds_offset(&atile[buf][0]), 32, 128, (long long)K);
    };

    issue_stage(0, 0);
    if (wave == 0) __builtin_amdgcn_s_wait_tensorcnt(0);
    wait_asynccnt0();
    __syncthreads();

    v8f c[2][4] = {{v8f{}, v8f{}, v8f{}, v8f{}}, {v8f{}, v8f{}, v8f{}, v8f{}}};
    int nsteps = K >> 5;
    for (int s = 0; s < nsteps; ++s) {
        int buf = s & 1;
        if (s + 1 < nsteps) issue_stage((s + 1) << 5, buf ^ 1);

        Frag16 a[2];
        #pragma unroll
        for (int mb = 0; mb < 2; ++mb) {
            const __bf16* ar = &atile[buf][(mw + mb * 16 + lo) * 32];
            #pragma unroll
            for (int v = 0; v < 8; ++v)
                a[mb].u[v] = *(const unsigned*)(ar + a_kidx(v, hi));
        }
        #pragma unroll
        for (int f = 0; f < 4; ++f) {
            const __bf16* br = &btile[buf][(f * 16 + lo) * 32 + 16 * hi];
            Frag16 b;
            #pragma unroll
            for (int v = 0; v < 8; ++v)
                b.u[v] = *(const unsigned*)(br + 2 * v);
            c[0][f] = wmma_bf16(a[0], b, c[0][f]);
            c[1][f] = wmma_bf16(a[1], b, c[1][f]);
        }
        if (s + 1 < nsteps) {
            if (wave == 0) __builtin_amdgcn_s_wait_tensorcnt(0);
            wait_asynccnt0();
        }
        __syncthreads();
    }

    #pragma unroll
    for (int mb = 0; mb < 2; ++mb)
        #pragma unroll
        for (int f = 0; f < 4; ++f) {
            int n = n0 + f * 16 + lo;
            if (n >= N) continue;
            #pragma unroll
            for (int v = 0; v < 8; ++v) {
                int m = row0 + mw + mb * 16 + v + 8 * hi;   // C row (ISA layout)
                float val = c[mb][f][v];
                if (EPI == EPI_UNEMB) {
                    ((float*)out)[(size_t)m * N + n] = val + extra[n];
                } else if (EPI == EPI_OPROJ) {
                    ((__bf16*)out)[(size_t)m * N + n] =
                        (__bf16)(val + extra[(size_t)m * N + n]);
                } else {
                    int b = m >> 11, t = m & (T_ - 1);
                    int h = n >> 6, dh = n & 63;
                    if (EPI == EPI_QHEAD)   // [B,H,T,Dh]
                        ((__bf16*)out)[((((size_t)b * H_ + h) * T_ + t) * DH_) + dh] =
                            (__bf16)val;
                    else                    // EPI_VHEAD: transposed [B,H,Dh,T]
                        ((__bf16*)out)[((((size_t)b * H_ + h) * DH_ + dh) * T_) + t] =
                            (__bf16)val;
                }
            }
        }
}

// ---------------------------------------------------------------------------
// 4) Causal flash attention. One wave per 16-query tile of one (b,h).
//    q,k: bf16 [B,H,T,Dh]; vt: bf16 [B,H,Dh,T]; z: bf16 [B*T, H*Dh].
//    Mask applied only on tiles crossing the diagonal (wave-uniform branch).
// ---------------------------------------------------------------------------
__global__ __launch_bounds__(128) void attn_kernel(
    const __bf16* __restrict__ q, const __bf16* __restrict__ k,
    const __bf16* __restrict__ vt, __bf16* __restrict__ z) {
    __shared__ __bf16 pLds[4 * 512];                     // 16x32 P tile / wave
    int lane = threadIdx.x & 31, wave = threadIdx.x >> 5;
    int hi = lane >> 4, lo = lane & 15;
    int gw = blockIdx.x * 4 + wave;                      // 0 .. B*H*(T/16)-1
    int qt = gw & (T_ / 16 - 1);
    int bh = gw >> 7;                                    // b*H + h
    int q0 = qt * 16;
    const __bf16* qbase = q  + (size_t)bh * T_ * DH_;
    const __bf16* kbase = k  + (size_t)bh * T_ * DH_;
    const __bf16* vbase = vt + (size_t)bh * DH_ * T_;
    __bf16* pl = &pLds[wave * 512];

    Frag16 qa[2];
    const __bf16* qrow = qbase + (size_t)(q0 + lo) * DH_;
    #pragma unroll
    for (int ch = 0; ch < 2; ++ch)
        #pragma unroll
        for (int v = 0; v < 8; ++v)
            qa[ch].u[v] = *(const unsigned*)(qrow + ch * 32 + a_kidx(v, hi));

    v8f o[4] = {v8f{}, v8f{}, v8f{}, v8f{}};
    float mrow[8], lrow[8];
    #pragma unroll
    for (int v = 0; v < 8; ++v) { mrow[v] = -1e30f; lrow[v] = 0.f; }
    const float scale = 0.125f;                          // 1/sqrt(64)
    int kend = q0 + 15;

    for (int k0 = 0; k0 <= kend; k0 += 32) {
        v8f s0 = {}, s1 = {};
        {
            const __bf16* kr0 = kbase + (size_t)(k0 + lo) * DH_ + 16 * hi;
            const __bf16* kr1 = kbase + (size_t)(k0 + 16 + lo) * DH_ + 16 * hi;
            Frag16 b0, b1, b2, b3;
            #pragma unroll
            for (int v = 0; v < 8; ++v) {
                b0.u[v] = *(const unsigned*)(kr0 + 2 * v);
                b1.u[v] = *(const unsigned*)(kr0 + 32 + 2 * v);
                b2.u[v] = *(const unsigned*)(kr1 + 2 * v);
                b3.u[v] = *(const unsigned*)(kr1 + 32 + 2 * v);
            }
            s0 = wmma_bf16(qa[0], b0, s0);
            s0 = wmma_bf16(qa[1], b1, s0);
            s1 = wmma_bf16(qa[0], b2, s1);
            s1 = wmma_bf16(qa[1], b3, s1);
        }

        // Online softmax update; row = v + 8*hi, col = lane&15 (C layout).
        auto softmax_update = [&](bool masked) {
            #pragma unroll
            for (int v = 0; v < 8; ++v) {
                int r = q0 + v + 8 * hi;
                float a0 = s0[v] * scale;
                float a1 = s1[v] * scale;
                if (masked) {
                    if (k0 + lo > r)      a0 = -1e30f;
                    if (k0 + 16 + lo > r) a1 = -1e30f;
                }
                float mx = fmaxf(a0, a1);
                #pragma unroll
                for (int off = 8; off > 0; off >>= 1)
                    mx = fmaxf(mx, __shfl_xor(mx, off, 32));
                float nm = fmaxf(mrow[v], mx);
                float corr = __expf(mrow[v] - nm);
                float p0 = __expf(a0 - nm);
                float p1 = __expf(a1 - nm);
                float rs = p0 + p1;
                #pragma unroll
                for (int off = 8; off > 0; off >>= 1)
                    rs += __shfl_xor(rs, off, 32);
                lrow[v] = lrow[v] * corr + rs;
                mrow[v] = nm;
                #pragma unroll
                for (int f = 0; f < 4; ++f) o[f][v] *= corr;
                pl[(v + 8 * hi) * 32 + lo]      = (__bf16)p0;
                pl[(v + 8 * hi) * 32 + 16 + lo] = (__bf16)p1;
            }
        };
        if (k0 + 31 > q0) softmax_update(true);   // tile crosses the diagonal
        else              softmax_update(false);  // fully unmasked fast path

        __builtin_amdgcn_wave_barrier();                 // keep LDS st->ld order
        Frag16 pa;
        #pragma unroll
        for (int v = 0; v < 8; ++v)
            pa.u[v] = *(const unsigned*)(pl + lo * 32 + a_kidx(v, hi));
        #pragma unroll
        for (int f = 0; f < 4; ++f) {
            int dh = f * 16 + lo;
            const __bf16* vrow = vbase + (size_t)dh * T_ + k0 + 16 * hi;
            Frag16 vb;
            #pragma unroll
            for (int v = 0; v < 8; ++v)
                vb.u[v] = *(const unsigned*)(vrow + 2 * v);
            o[f] = wmma_bf16(pa, vb, o[f]);
        }
    }

    int b = bh >> 4, h = bh & 15;
    #pragma unroll
    for (int f = 0; f < 4; ++f)
        #pragma unroll
        for (int v = 0; v < 8; ++v) {
            int t = q0 + v + 8 * hi;
            float inv = 1.f / lrow[v];
            z[((size_t)(b * T_ + t)) * (H_ * DH_) + h * DH_ + f * 16 + lo] =
                (__bf16)(o[f][v] * inv);
        }
}

// ---------------------------------------------------------------------------
// Launch
// ---------------------------------------------------------------------------
extern "C" void kernel_launch(void* const* d_in, const int* in_sizes, int n_in,
                              void* d_out, int out_size, void* d_ws, size_t ws_size,
                              hipStream_t stream) {
    (void)in_sizes; (void)n_in; (void)out_size; (void)ws_size;
    const int*   x     = (const int*)d_in[0];
    const float* W_E   = (const float*)d_in[1];
    const float* W_pos = (const float*)d_in[2];
    const float* W_Q   = (const float*)d_in[3];
    const float* W_K   = (const float*)d_in[4];
    const float* W_V   = (const float*)d_in[5];
    const float* W_O   = (const float*)d_in[6];
    const float* W_U   = (const float*)d_in[7];
    const float* b_U   = (const float*)d_in[8];
    float* out = (float*)d_out;

    char* ws = (char*)d_ws;
    size_t off = 0;
    auto alloc = [&](size_t bytes) -> void* {
        void* p = ws + off;
        off = (off + bytes + 255) & ~(size_t)255;
        return p;
    };
    float*  hf    = (float*)alloc((size_t)M_ * D_ * 4);
    __bf16* hb    = (__bf16*)alloc((size_t)M_ * D_ * 2);
    __bf16* qb    = (__bf16*)alloc((size_t)B_ * H_ * T_ * DH_ * 2);
    __bf16* kb    = (__bf16*)alloc((size_t)B_ * H_ * T_ * DH_ * 2);
    __bf16* vtb   = (__bf16*)alloc((size_t)B_ * H_ * DH_ * T_ * 2);
    __bf16* zb    = (__bf16*)alloc((size_t)M_ * D_ * 2);
    __bf16* hidb  = (__bf16*)alloc((size_t)M_ * D_ * 2);
    __bf16* wq_t  = (__bf16*)alloc((size_t)D_ * D_ * 2);
    __bf16* wk_t  = (__bf16*)alloc((size_t)D_ * D_ * 2);
    __bf16* wv_t  = (__bf16*)alloc((size_t)D_ * D_ * 2);
    __bf16* wo_t  = (__bf16*)alloc((size_t)D_ * D_ * 2);
    __bf16* wu_t  = (__bf16*)alloc((size_t)V_ * D_ * 2);

    dim3 tb(32, 8);
    dim3 tg1((D_ + 31) / 32, (D_ + 31) / 32);
    transpose_convert_kernel<<<tg1, tb, 0, stream>>>(W_Q, wq_t, D_, D_);
    transpose_convert_kernel<<<tg1, tb, 0, stream>>>(W_K, wk_t, D_, D_);
    transpose_convert_kernel<<<tg1, tb, 0, stream>>>(W_V, wv_t, D_, D_);
    transpose_convert_kernel<<<tg1, tb, 0, stream>>>(W_O, wo_t, D_, D_);
    transpose_convert_kernel<<<dim3((V_ + 31) / 32, (D_ + 31) / 32), tb, 0,
                               stream>>>(W_U, wu_t, D_, V_);

    embed_kernel<<<(M_ * D_) / 256, 256, 0, stream>>>(x, W_E, W_pos, hf, hb);

    dim3 gproj(D_ / 64, M_ / 128);
    gemm_kernel<EPI_QHEAD><<<gproj, 128, 0, stream>>>(hb, wq_t, M_, D_, D_, qb, nullptr);
    gemm_kernel<EPI_QHEAD><<<gproj, 128, 0, stream>>>(hb, wk_t, M_, D_, D_, kb, nullptr);
    gemm_kernel<EPI_VHEAD><<<gproj, 128, 0, stream>>>(hb, wv_t, M_, D_, D_, vtb, nullptr);

    attn_kernel<<<(B_ * H_ * (T_ / 16)) / 4, 128, 0, stream>>>(qb, kb, vtb, zb);

    gemm_kernel<EPI_OPROJ><<<gproj, 128, 0, stream>>>(zb, wo_t, M_, D_, D_, hidb, hf);

    dim3 gunemb((V_ + 63) / 64, M_ / 128);
    gemm_kernel<EPI_UNEMB><<<gunemb, 128, 0, stream>>>(hidb, wu_t, M_, V_, D_, out, b_U);
}
